// MoE_11235634446828
// MI455X (gfx1250) — matrix operand order repaired and checked by reading, not compiled, not run on hardware
//
#include <hip/hip_runtime.h>
#include <hip/hip_bf16.h>
#include <stdint.h>

// Problem dims (fixed by reference)
#define BROWS 4096
#define DDIM  512
#define HDIM  1024
#define ODIM  512
#define ENUM  16

typedef __attribute__((ext_vector_type(16))) __bf16 v16bf;
typedef __attribute__((ext_vector_type(8)))  float  v8f;

union Frag {
  uint4 u4[2];   // 32 bytes = 8 dwords = 16 bf16
  v16bf v;
};

__device__ __forceinline__ unsigned short f32_to_bf16(float f) {
  unsigned int u = __float_as_uint(f);
  u += 0x7FFFu + ((u >> 16) & 1u);   // round-to-nearest-even
  return (unsigned short)(u >> 16);
}

// ---------------------------------------------------------------- stats zero
__global__ void zero_stats_kernel(float* __restrict__ stats) {
  if (threadIdx.x < 32) stats[threadIdx.x] = 0.0f;
}

// ---------------------------------------------------------------- gating
// One wave (32 threads) per row. Lanes 0..15 each compute one expert logit
// (w_gate/task_gate reads coalesced across lanes). Lane 0 does top-4 + softmax.
__global__ __launch_bounds__(32) void gating_kernel(
    const float* __restrict__ query, const float* __restrict__ x,
    const float* __restrict__ wg, const float* __restrict__ tg,
    float* __restrict__ gates, float* __restrict__ imp, float* __restrict__ loadc) {
  __shared__ float lg[ENUM];
  __shared__ float gout[ENUM];
  const int row  = blockIdx.x;
  const int lane = threadIdx.x;
  if (lane < ENUM) {
    float acc = 0.0f;
    const float* xr = x + (size_t)row * DDIM;
    for (int d = 0; d < DDIM; ++d) acc += xr[d] * wg[d * ENUM + lane];
    const float* qr = query + (size_t)row * 256;
    for (int q = 0; q < 256; ++q) acc += qr[q] * tg[q * ENUM + lane];
    lg[lane]   = acc;
    gout[lane] = 0.0f;
  }
  __syncthreads();
  if (lane == 0) {
    float v[ENUM]; int used[ENUM];
    for (int i = 0; i < ENUM; ++i) { v[i] = lg[i]; used[i] = 0; }
    int idx[4]; float val[4];
    for (int t = 0; t < 4; ++t) {
      int bi = 0; float bv = -3.4e38f;
      for (int i = 0; i < ENUM; ++i)
        if (!used[i] && v[i] > bv) { bv = v[i]; bi = i; }
      used[bi] = 1; idx[t] = bi; val[t] = bv;
    }
    const float mx = val[0];
    float ex[4], s = 0.0f;
    for (int t = 0; t < 4; ++t) { ex[t] = __expf(val[t] - mx); s += ex[t]; }
    const float inv = 1.0f / s;
    for (int t = 0; t < 4; ++t) {
      const float g = ex[t] * inv;
      gout[idx[t]] = g;
      atomicAdd(&imp[idx[t]], g);
      atomicAdd(&loadc[idx[t]], 1.0f);
    }
  }
  __syncthreads();
  if (lane < ENUM) gates[(size_t)row * ENUM + lane] = gout[lane];
}

// ---------------------------------------------------------------- loss
__global__ void loss_kernel(const float* __restrict__ imp,
                            const float* __restrict__ loadc,
                            float* __restrict__ out) {
  if (threadIdx.x == 0) {
    float mi = 0.f, ml = 0.f;
    for (int i = 0; i < ENUM; ++i) { mi += imp[i]; ml += loadc[i]; }
    mi *= (1.0f / ENUM); ml *= (1.0f / ENUM);
    float vi = 0.f, vl = 0.f;
    for (int i = 0; i < ENUM; ++i) {
      float a = imp[i] - mi;  vi += a * a;
      float b = loadc[i] - ml; vl += b * b;
    }
    vi *= (1.0f / (ENUM - 1)); vl *= (1.0f / (ENUM - 1));
    out[0] = 0.01f * (vi / (mi * mi + 1e-10f) + vl / (ml * ml + 1e-10f));
  }
}

// ---------------------------------------------------------------- conversions
__global__ void cvt_x_kernel(const float* __restrict__ x,
                             unsigned short* __restrict__ xbf, int n) {
  int i = blockIdx.x * blockDim.x + threadIdx.x;
  if (i < n) xbf[i] = f32_to_bf16(x[i]);
}

// W1 [E][D][H] -> w1t bf16 [E][H][D]
__global__ void cvt_w1t_kernel(const float* __restrict__ W1,
                               unsigned short* __restrict__ w1t) {
  int i = blockIdx.x * blockDim.x + threadIdx.x;   // over E*H*D = 8388608
  int d = i & (DDIM - 1);
  int h = (i >> 9) & (HDIM - 1);
  int e = i >> 19;
  w1t[i] = f32_to_bf16(W1[((size_t)e * DDIM + d) * HDIM + h]);
}

// W2 [E][H][O] -> w2t bf16 [E][O][H]
__global__ void cvt_w2t_kernel(const float* __restrict__ W2,
                               unsigned short* __restrict__ w2t) {
  int i = blockIdx.x * blockDim.x + threadIdx.x;   // over E*O*H = 8388608
  int h = i & (HDIM - 1);
  int o = (i >> 10) & (ODIM - 1);
  int e = i >> 19;
  w2t[i] = f32_to_bf16(W2[((size_t)e * HDIM + h) * ODIM + o]);
}

// ---------------------------------------------------------------- fused MoE MLP
// Block: 256 threads (8 waves). Block tile: 32 rows x 512 output cols.
// Each wave: 64 output cols (GEMM2) and one 16-wide h slice (GEMM1); two
// 16-row m-tiles share every B fragment (2 WMMAs per weight load). Loop over
// all 16 experts; the gate is folded into the ReLU'd h tile so one persistent
// fp32 accumulator sums all experts without atomics.
__global__ __launch_bounds__(256) void moe_mlp_kernel(
    const float* __restrict__ b1,             // [E][H]
    const float* __restrict__ b2,             // [E][O]
    const unsigned short* __restrict__ xbf,   // bf16 [B][D]
    const unsigned short* __restrict__ w1t,   // bf16 [E][H][D]
    const unsigned short* __restrict__ w2t,   // bf16 [E][O][H]
    const float* __restrict__ gates,          // [B][E]
    float* __restrict__ y) {                  // [B][O]
  __shared__ unsigned short xtile[32 * DDIM]; // 32 KB
  __shared__ unsigned short htile[32 * 128];  // 8 KB (one 128-wide H chunk)
  __shared__ float gtile[32 * ENUM];          // 2 KB

  const int tid     = threadIdx.x;
  const int wave    = tid >> 5;
  const int lane    = tid & 31;
  const int rowbase = blockIdx.x * 32;
  const int n16     = lane & 15;   // row (A) / col (B,D) index within 16-tile
  const int khalf   = lane >> 4;   // K-half selector: this code's A/B order is k = 16*khalf + s
  const int colbase = wave * 64;   // this wave's output columns

  { // stage x tile (32 rows x 512 bf16) into LDS, reused for all experts
    const uint4* src = (const uint4*)(xbf + (size_t)rowbase * DDIM);
    uint4* dst = (uint4*)xtile;
    for (int i = tid; i < 2048; i += 256) dst[i] = src[i];
  }
  for (int i = tid; i < 512; i += 256) gtile[i] = gates[(size_t)rowbase * ENUM + i];
  __syncthreads();

  const v8f vzero = {0.f, 0.f, 0.f, 0.f, 0.f, 0.f, 0.f, 0.f};
  v8f yacc[2][4];
#pragma unroll
  for (int mt = 0; mt < 2; ++mt)
#pragma unroll
    for (int t = 0; t < 4; ++t) yacc[mt][t] = vzero;

  for (int e = 0; e < ENUM; ++e) {
    float gm[2][8];          // gate per m for D-layout rows m = v + 8*khalf
#pragma unroll
    for (int mt = 0; mt < 2; ++mt)
#pragma unroll
      for (int v = 0; v < 8; ++v)
        gm[mt][v] = gtile[(mt * 16 + v + 8 * khalf) * ENUM + e];

    for (int outer = 0; outer < 8; ++outer) {      // H in chunks of 128
      const int hc = outer * 128 + wave * 16;      // this wave's 16 h-cols
      // ---- GEMM1: h[32 x 16] = x[32 x 512] @ W1[512 x 16]
      v8f hacc[2] = {vzero, vzero};
      const unsigned short* pB = w1t + ((size_t)(e * HDIM + hc + n16)) * DDIM;
#pragma unroll 4
      for (int ks = 0; ks < 16; ++ks) {
        const int kb = ks * 32;
        Frag b;
        const uint4* q = (const uint4*)(pB + kb + khalf * 16);
        b.u4[0] = q[0]; b.u4[1] = q[1];
#pragma unroll
        for (int mt = 0; mt < 2; ++mt) {
          Frag a;
          const unsigned short* pa =
              xtile + (mt * 16 + n16) * DDIM + kb + khalf * 16;
          a.u4[0] = *(const uint4*)pa;
          a.u4[1] = *(const uint4*)(pa + 8);
          hacc[mt] = __builtin_amdgcn_wmma_f32_16x16x32_bf16(
              false, a.v, false, b.v, (short)0, hacc[mt], false, false);
        }
      }
      // bias + ReLU + gate-scale -> bf16 h tile in LDS
      const float b1v = b1[e * HDIM + hc + n16];
#pragma unroll
      for (int mt = 0; mt < 2; ++mt)
#pragma unroll
        for (int v = 0; v < 8; ++v) {
          float h = hacc[mt][v] + b1v;
          h = h > 0.f ? h : 0.f;
          htile[(mt * 16 + v + 8 * khalf) * 128 + wave * 16 + n16] =
              f32_to_bf16(h * gm[mt][v]);
        }
      __syncthreads();
      // ---- GEMM2: yacc += h[32 x 128] @ W2[128 x 64(this wave)]
#pragma unroll
      for (int ks2 = 0; ks2 < 4; ++ks2) {
        Frag a2[2];
#pragma unroll
        for (int mt = 0; mt < 2; ++mt) {
          const unsigned short* pa2 =
              htile + (mt * 16 + n16) * 128 + ks2 * 32 + khalf * 16;
          a2[mt].u4[0] = *(const uint4*)pa2;
          a2[mt].u4[1] = *(const uint4*)(pa2 + 8);
        }
#pragma unroll
        for (int t = 0; t < 4; ++t) {
          Frag bf;
          const unsigned short* pb =
              w2t + ((size_t)(e * ODIM + colbase + t * 16 + n16)) * HDIM +
              outer * 128 + ks2 * 32 + khalf * 16;
          bf.u4[0] = ((const uint4*)pb)[0];
          bf.u4[1] = ((const uint4*)pb)[1];
#pragma unroll
          for (int mt = 0; mt < 2; ++mt)
            yacc[mt][t] = __builtin_amdgcn_wmma_f32_16x16x32_bf16(
                false, a2[mt].v, false, bf.v, (short)0, yacc[mt][t], false, false);
        }
      }
      __syncthreads();   // htile reused next chunk
    }
  }

  // ---- epilogue: y = yacc + sum_e gate[m,e] * b2[e][j]
#pragma unroll
  for (int mt = 0; mt < 2; ++mt)
#pragma unroll
    for (int t = 0; t < 4; ++t) {
      const int j = colbase + t * 16 + n16;
      float bias[8];
#pragma unroll
      for (int v = 0; v < 8; ++v) bias[v] = 0.f;
      for (int e = 0; e < ENUM; ++e) {
        const float b2v = b2[e * ODIM + j];
#pragma unroll
        for (int v = 0; v < 8; ++v)
          bias[v] += gtile[(mt * 16 + v + 8 * khalf) * ENUM + e] * b2v;
      }
#pragma unroll
      for (int v = 0; v < 8; ++v) {
        const int m = mt * 16 + v + 8 * khalf;
        y[(size_t)(rowbase + m) * ODIM + j] = yacc[mt][t][v] + bias[v];
      }
    }
}

// ---------------------------------------------------------------- launch
extern "C" void kernel_launch(void* const* d_in, const int* in_sizes, int n_in,
                              void* d_out, int out_size, void* d_ws, size_t ws_size,
                              hipStream_t stream) {
  const float* query = (const float*)d_in[0];   // [B,256]
  const float* x     = (const float*)d_in[1];   // [B,512]
  const float* wg    = (const float*)d_in[2];   // [512,16]
  const float* tg    = (const float*)d_in[3];   // [256,16]
  const float* W1    = (const float*)d_in[4];   // [16,512,1024]
  const float* b1    = (const float*)d_in[5];   // [16,1024]
  const float* W2    = (const float*)d_in[6];   // [16,1024,512]
  const float* b2    = (const float*)d_in[7];   // [16,512]
  float* y = (float*)d_out;                     // [B,512] then loss at [B*512]

  char* ws = (char*)d_ws;
  float* gates = (float*)(ws + 0);                                 // 256 KB
  float* stats = (float*)(ws + 262144);                            // imp[16], load[16]
  unsigned short* xbf = (unsigned short*)(ws + 262400);            // 4 MB
  unsigned short* w1t = (unsigned short*)(ws + 262400 + 4194304);  // 16 MB
  unsigned short* w2t = (unsigned short*)(ws + 262400 + 4194304 + 16777216);

  zero_stats_kernel<<<1, 32, 0, stream>>>(stats);
  gating_kernel<<<BROWS, 32, 0, stream>>>(query, x, wg, tg, gates, stats, stats + 16);
  loss_kernel<<<1, 32, 0, stream>>>(stats, stats + 16, y + (size_t)BROWS * ODIM);
  cvt_x_kernel<<<(BROWS * DDIM) / 256, 256, 0, stream>>>(x, xbf, BROWS * DDIM);
  cvt_w1t_kernel<<<(ENUM * HDIM * DDIM) / 256, 256, 0, stream>>>(W1, w1t);
  cvt_w2t_kernel<<<(ENUM * ODIM * HDIM) / 256, 256, 0, stream>>>(W2, w2t);
  moe_mlp_kernel<<<BROWS / 32, 256, 0, stream>>>(b1, b2, xbf, w1t, w2t, gates, y);
}
